// Attention_8486855377309
// MI455X (gfx1250) — matrix operand order repaired and checked by reading, not compiled, not run on hardware
//
#include <hip/hip_runtime.h>
#include <hip/hip_bf16.h>

// ---------------------------------------------------------------------------
// Causal multi-head attention for MI455X (gfx1250), bf16 WMMA pipeline.
// B=2, S=2048, D_MODEL=1024, H=16, D_HEAD=64.
// Software-pipelined loads + LDS weight staging via gfx1250 async
// global->LDS copies (ASYNCcnt) with a portable fallback.
// ---------------------------------------------------------------------------

typedef __bf16 bf16_t;
typedef __attribute__((ext_vector_type(16))) __bf16 v16bf;
typedef __attribute__((ext_vector_type(8)))  __bf16 v8bf;
typedef __attribute__((ext_vector_type(8)))  float  v8f;
typedef int v4i __attribute__((vector_size(16)));  // matches builtin param type

#define D_MODEL 1024
#define N_HEADS 16
#define D_HEAD  64
#define SEQ     2048
#define BATCH   2
#define NTOK    (BATCH * SEQ)
#define IGNORE_VAL (-100000.0f)

// LDS staging geometry: 64 rows x 256 k-slice, padded row stride 264 elems
// (528 B = 132 dwords -> bank advance 4/row -> the 16 B-tile lanes land on
// distinct banks).
#define KSLICE   256
#define SROWS    64
#define SSTRIDE  264

#if defined(__has_builtin)
#if __has_builtin(__builtin_amdgcn_global_load_async_to_lds_b128) && \
    __has_builtin(__builtin_amdgcn_s_wait_asynccnt)
#define HAVE_ASYNC_LDS 1
#endif
#endif

// ---------------------------------------------------------------------------
// WMMA helper: D = A(16x32 bf16) * B(32x16 bf16) + C(16x16 f32)
// ---------------------------------------------------------------------------
__device__ __forceinline__ v8f wmma_bf16(v16bf a, v16bf b, v8f c) {
  return __builtin_amdgcn_wmma_f32_16x16x32_bf16(
      /*neg_a=*/false, a, /*neg_b=*/false, b,
      /*c_mod=*/(short)0, c, /*reuse_a=*/false, /*reuse_b=*/false);
}

// A-matrix 16x32 (MxK) tile load from row-major [row][ld] bf16 storage.
// Contiguous k order used here: lane m = l&15; elem e -> K = half*16 + e
// (half = l>>4). Two contiguous 16B loads per lane.
__device__ __forceinline__ v16bf load_a_tile(const bf16_t* __restrict__ base,
                                             int ld, int row0, int k0, int lane) {
  int m = lane & 15, half = lane >> 4;
  const bf16_t* p = base + (size_t)(row0 + m) * ld + k0 + half * 16;
  v8bf lo = *(const v8bf*)p;
  v8bf hi = *(const v8bf*)(p + 8);
  v16bf a;
#pragma unroll
  for (int i = 0; i < 8; ++i) { a[i] = lo[i]; a[i + 8] = hi[i]; }
  return a;
}

// B-matrix 32x16 (KxN) tile load. Source stored TRANSPOSED: element (k,n) at
// base[n*ld + k]. Contiguous k order used here: lane n = l&15 fixed column, elem e -> K =
// half*16 + e. Two 16B loads per lane (works for global or LDS pointers; for
// LDS the padded stride keeps 16B alignment and conflict-free banking).
__device__ __forceinline__ v16bf load_b_tile(const bf16_t* __restrict__ base,
                                             int ld, int n0, int k0, int lane) {
  int n = lane & 15, half = lane >> 4;
  const bf16_t* p = base + (size_t)(n0 + n) * ld + k0 + half * 16;
  v8bf lo = *(const v8bf*)p;
  v8bf hi = *(const v8bf*)(p + 8);
  v16bf b;
#pragma unroll
  for (int i = 0; i < 8; ++i) { b[i] = lo[i]; b[i + 8] = hi[i]; }
  return b;
}

// ---------------------------------------------------------------------------
// Stage a 64 x 256 bf16 weight panel (global row stride gld) into LDS with
// padded stride SSTRIDE. Uses gfx1250 async global->LDS DMA when available.
// ---------------------------------------------------------------------------
template <int NTHREADS>
__device__ __forceinline__ void stage_w_panel(const bf16_t* __restrict__ g,
                                              int gld, bf16_t* s, int tid) {
  // 64 rows * 32 chunks of 16B = 2048 chunks
#pragma unroll
  for (int c = 0; c < (SROWS * 32); c += NTHREADS) {
    int idx = c + tid;
    int row = idx >> 5;
    int col = (idx & 31) * 8;
    const bf16_t* gp = g + (size_t)row * gld + col;
    bf16_t* lp = s + row * SSTRIDE + col;
#if HAVE_ASYNC_LDS
    __builtin_amdgcn_global_load_async_to_lds_b128(
        (__attribute__((address_space(1))) v4i*)gp,
        (__attribute__((address_space(3))) v4i*)lp, 0, 0);
#else
    *(v8bf*)lp = *(const v8bf*)gp;
#endif
  }
#if HAVE_ASYNC_LDS
  __builtin_amdgcn_s_wait_asynccnt(0);
#endif
  __syncthreads();
}

// ---------------------------------------------------------------------------
// Conversion kernels
// ---------------------------------------------------------------------------
__global__ void convert_f32_bf16(const float* __restrict__ src,
                                 bf16_t* __restrict__ dst, int n) {
  int i = blockIdx.x * blockDim.x + threadIdx.x;
  if (i < n) dst[i] = (bf16_t)src[i];
}

// src [batch][R][C] f32 -> dst [batch][C][R] bf16
__global__ void transpose_convert(const float* __restrict__ src,
                                  bf16_t* __restrict__ dst,
                                  int batch, int R, int C) {
  int i = blockIdx.x * blockDim.x + threadIdx.x;
  int total = batch * R * C;
  if (i >= total) return;
  int c = i % C;
  int r = (i / C) % R;
  int b = i / (R * C);
  dst[(size_t)b * R * C + (size_t)c * R + r] = (bf16_t)src[i];
}

// ---------------------------------------------------------------------------
// QKV projection: per head h, per matrix (Q/K/V):
//   out[token][e] = sum_d x[token][d] * W[h][d][e] + bias[h][e]
// A = x (bf16 [4096][1024]) streamed from global with register double-buffer;
// B = W^T panel staged in LDS (shared by all 4 waves).
// Q,K stored row-major [b][h][s][64]; V stored transposed [b][h][64][s].
// ---------------------------------------------------------------------------
__global__ __launch_bounds__(128) void qkv_kernel(
    const bf16_t* __restrict__ xb,
    const bf16_t* __restrict__ wqt, const bf16_t* __restrict__ wkt,
    const bf16_t* __restrict__ wvt,
    const float* __restrict__ bq, const float* __restrict__ bk,
    const float* __restrict__ bv,
    bf16_t* __restrict__ Qb, bf16_t* __restrict__ Kb,
    bf16_t* __restrict__ Vt) {
  __shared__ bf16_t sW[SROWS * SSTRIDE];

  int lane = threadIdx.x & 31;
  int wave = threadIdx.x >> 5;
  int row0 = blockIdx.x * 64 + wave * 16;   // token row of this wave's strip
  int h    = blockIdx.y;
  int mat  = blockIdx.z;                    // 0=Q 1=K 2=V

  const bf16_t* w    = (mat == 0) ? wqt : (mat == 1) ? wkt : wvt;
  const float*  bias = (mat == 0) ? bq  : (mat == 1) ? bk  : bv;
  const bf16_t* wh   = w + (size_t)h * D_HEAD * D_MODEL;

  v8f acc[4] = {v8f{}, v8f{}, v8f{}, v8f{}};
  for (int ks = 0; ks < D_MODEL; ks += KSLICE) {
    stage_w_panel<128>(wh + ks, D_MODEL, sW, threadIdx.x);

    v16bf a_cur = load_a_tile(xb, D_MODEL, row0, ks, lane);
#pragma unroll
    for (int k0 = 0; k0 < KSLICE; k0 += 32) {
      v16bf a_next;
      if (k0 + 32 < KSLICE)  // prefetch next A tile before consuming current
        a_next = load_a_tile(xb, D_MODEL, row0, ks + k0 + 32, lane);
#pragma unroll
      for (int nt = 0; nt < 4; ++nt) {
        v16bf b = load_b_tile(sW, SSTRIDE, nt * 16, k0, lane);
        acc[nt] = wmma_bf16(a_cur, b, acc[nt]);
      }
      if (k0 + 32 < KSLICE) a_cur = a_next;
    }
    __syncthreads();  // protect sW before restaging
  }

  int n = lane & 15, half = lane >> 4;
  int bidx = row0 >> 11;       // batch (strips never cross batch: 2048 % 64 == 0)
  int s0   = row0 & (SEQ - 1);
#pragma unroll
  for (int nt = 0; nt < 4; ++nt) {
    int e = nt * 16 + n;
    float bs = bias[h * D_HEAD + e];
    if (mat < 2) {
      bf16_t* outp = ((mat == 0) ? Qb : Kb) +
                     ((size_t)(bidx * N_HEADS + h)) * SEQ * D_HEAD;
#pragma unroll
      for (int r = 0; r < 8; ++r) {
        int s = s0 + r + 8 * half;
        outp[(size_t)s * D_HEAD + e] = (bf16_t)(acc[nt][r] + bs);
      }
    } else {
      // V transposed: consecutive regs -> consecutive s -> one packed 16B store
      v8bf pk;
#pragma unroll
      for (int r = 0; r < 8; ++r) pk[r] = (bf16_t)(acc[nt][r] + bs);
      bf16_t* outp = Vt + ((size_t)(bidx * N_HEADS + h) * D_HEAD + e) * SEQ +
                     s0 + 8 * half;
      *(v8bf*)outp = pk;
    }
  }
}

// ---------------------------------------------------------------------------
// Flash attention per (b, h, 16-row q strip). Computes scores^T = K * Q^T so
// the C-tile layout puts each q-row in a fixed lane: softmax reductions are
// lane-local + one shfl_xor(16); z^T = V^T * P^T with P^T assembled by 16
// shuffles (no LDS). Pipelined: V tiles issued before the score WMMAs, next
// K tiles prefetched before the softmax math.
// ---------------------------------------------------------------------------
__global__ __launch_bounds__(128) void attn_kernel(
    const bf16_t* __restrict__ Qb, const bf16_t* __restrict__ Kb,
    const bf16_t* __restrict__ Vt, bf16_t* __restrict__ Zb) {
  int lane = threadIdx.x & 31;
  int wave = threadIdx.x >> 5;
  int q0 = blockIdx.x * 64 + wave * 16;  // first q row of this wave
  int h  = blockIdx.y;
  int b  = blockIdx.z;

  const bf16_t* Qh = Qb + (size_t)(b * N_HEADS + h) * SEQ * D_HEAD;
  const bf16_t* Kh = Kb + (size_t)(b * N_HEADS + h) * SEQ * D_HEAD;
  const bf16_t* Vh = Vt + (size_t)(b * N_HEADS + h) * D_HEAD * SEQ;

  int n = lane & 15, half = lane >> 4;
  int q = q0 + n;  // the q-row this lane owns in every C tile

  // B = Q^T tiles (k = e), loaded once, reused for all key blocks
  v16bf bq0 = load_b_tile(Qh, D_HEAD, q0, 0, lane);
  v16bf bq1 = load_b_tile(Qh, D_HEAD, q0, 32, lane);

  v8f accz[4] = {v8f{}, v8f{}, v8f{}, v8f{}};  // z^T: e = eb*16 + r + 8*half
  float m_run = -3.0e38f;
  float l_run = 0.0f;
  const float scale = 0.125f;  // 1/sqrt(64)
  const int kend = q0 + 15;

  // preload K tiles for first key block
  v16bf ak0 = load_a_tile(Kh, D_HEAD, 0,  0,  lane);
  v16bf ak1 = load_a_tile(Kh, D_HEAD, 0,  32, lane);
  v16bf ak2 = load_a_tile(Kh, D_HEAD, 16, 0,  lane);
  v16bf ak3 = load_a_tile(Kh, D_HEAD, 16, 32, lane);

  for (int kb = 0; kb <= kend; kb += 32) {
    // issue V-tile loads first so the score WMMAs wait only on older K loads
    v16bf av0 = load_a_tile(Vh, SEQ, 0,  kb, lane);
    v16bf av1 = load_a_tile(Vh, SEQ, 16, kb, lane);
    v16bf av2 = load_a_tile(Vh, SEQ, 32, kb, lane);
    v16bf av3 = load_a_tile(Vh, SEQ, 48, kb, lane);

    // scores^T: S0 covers keys kb..kb+15, S1 covers kb+16..kb+31
    v8f s0 = v8f{};
    s0 = wmma_bf16(ak0, bq0, s0);
    s0 = wmma_bf16(ak1, bq1, s0);
    v8f s1 = v8f{};
    s1 = wmma_bf16(ak2, bq0, s1);
    s1 = wmma_bf16(ak3, bq1, s1);

    // prefetch next block's K tiles; they fly during the softmax math
    bool more = (kb + 32) <= kend;  // wave-uniform
    v16bf an0, an1, an2, an3;
    if (more) {
      an0 = load_a_tile(Kh, D_HEAD, kb + 32, 0,  lane);
      an1 = load_a_tile(Kh, D_HEAD, kb + 32, 32, lane);
      an2 = load_a_tile(Kh, D_HEAD, kb + 48, 0,  lane);
      an3 = load_a_tile(Kh, D_HEAD, kb + 48, 32, lane);
    }

    // scale + causal mask; per-lane block max over this lane's 16 keys
    float p0[8], p1[8];
    float blkmax = -3.0e38f;
#pragma unroll
    for (int r = 0; r < 8; ++r) {
      int key0 = kb + r + 8 * half;
      int key1 = key0 + 16;
      float v0 = (key0 <= q) ? s0[r] * scale : IGNORE_VAL;
      float v1 = (key1 <= q) ? s1[r] * scale : IGNORE_VAL;
      p0[r] = v0; p1[r] = v1;
      blkmax = fmaxf(blkmax, fmaxf(v0, v1));
    }
    // combine the two lanes holding the same q-row (half 0 / half 1)
    blkmax = fmaxf(blkmax, __shfl_xor(blkmax, 16, 32));
    float m_new = fmaxf(m_run, blkmax);
    float corr = __expf(m_run - m_new);

    float lsum = 0.0f;
#pragma unroll
    for (int r = 0; r < 8; ++r) {
      p0[r] = __expf(p0[r] - m_new);
      p1[r] = __expf(p1[r] - m_new);
      lsum += p0[r] + p1[r];
    }
    lsum += __shfl_xor(lsum, 16, 32);
    l_run = l_run * corr + lsum;
    m_run = m_new;

#pragma unroll
    for (int t = 0; t < 4; ++t)
#pragma unroll
      for (int r = 0; r < 8; ++r) accz[t][r] *= corr;

    // Assemble B = P^T (k = local key 0..31, n = q-row) via cross-half shfl:
    //   half0 lane needs keys 0..15 ; half1 lane needs keys 16..31
    v16bf bp;
#pragma unroll
    for (int r = 0; r < 8; ++r) {
      float x0 = __shfl_xor(p0[r], 16, 32);  // partner's S0 value
      float x1 = __shfl_xor(p1[r], 16, 32);  // partner's S1 value
      float e_lo = (half == 0) ? p0[r] : x1;   // keys r      / 16+r
      float e_hi = (half == 0) ? x0    : p1[r];// keys 8+r    / 24+r
      bp[r]     = (bf16_t)e_lo;
      bp[r + 8] = (bf16_t)e_hi;
    }

    // z^T += V^T_tile * P^T for 4 e-blocks (V tiles were issued first)
    accz[0] = wmma_bf16(av0, bp, accz[0]);
    accz[1] = wmma_bf16(av1, bp, accz[1]);
    accz[2] = wmma_bf16(av2, bp, accz[2]);
    accz[3] = wmma_bf16(av3, bp, accz[3]);

    if (more) { ak0 = an0; ak1 = an1; ak2 = an2; ak3 = an3; }
  }

  // finalize: divide by softmax denominator (lane-local, row = n) and store
  float inv = 1.0f / l_run;
  int tok = b * SEQ + q;
  bf16_t* zrow = Zb + (size_t)tok * D_MODEL + h * D_HEAD;
#pragma unroll
  for (int t = 0; t < 4; ++t)
#pragma unroll
    for (int r = 0; r < 8; ++r) {
      int e = t * 16 + r + 8 * half;
      zrow[e] = (bf16_t)(accz[t][r] * inv);
    }
}

// ---------------------------------------------------------------------------
// Output projection: out[token][d] = sum_f z[token][f] * W_O[f][d] + b_O[d]
// A = Zb bf16 [4096][1024] (register double-buffered);
// B = W_O^T panel staged in LDS (shared by all 8 waves).
// Block: 256 threads = 8 waves; block tile 128(M) x 64(N).
// ---------------------------------------------------------------------------
__global__ __launch_bounds__(256) void oproj_kernel(
    const bf16_t* __restrict__ Zb, const bf16_t* __restrict__ wot,
    const float* __restrict__ bo, float* __restrict__ out) {
  __shared__ bf16_t sW[SROWS * SSTRIDE];

  int lane = threadIdx.x & 31;
  int wave = threadIdx.x >> 5;
  int row0 = blockIdx.x * 128 + wave * 16;
  int n0   = blockIdx.y * 64;

  v8f acc[4] = {v8f{}, v8f{}, v8f{}, v8f{}};
  for (int ks = 0; ks < D_MODEL; ks += KSLICE) {
    stage_w_panel<256>(wot + (size_t)n0 * D_MODEL + ks, D_MODEL, sW, threadIdx.x);

    v16bf a_cur = load_a_tile(Zb, D_MODEL, row0, ks, lane);
#pragma unroll
    for (int k0 = 0; k0 < KSLICE; k0 += 32) {
      v16bf a_next;
      if (k0 + 32 < KSLICE)
        a_next = load_a_tile(Zb, D_MODEL, row0, ks + k0 + 32, lane);
#pragma unroll
      for (int nt = 0; nt < 4; ++nt) {
        v16bf b = load_b_tile(sW, SSTRIDE, nt * 16, k0, lane);
        acc[nt] = wmma_bf16(a_cur, b, acc[nt]);
      }
      if (k0 + 32 < KSLICE) a_cur = a_next;
    }
    __syncthreads();
  }

  int n = lane & 15, half = lane >> 4;
#pragma unroll
  for (int nt = 0; nt < 4; ++nt) {
    int col = n0 + nt * 16 + n;
    float bias = bo[col];
#pragma unroll
    for (int r = 0; r < 8; ++r) {
      int row = row0 + r + 8 * half;
      out[(size_t)row * D_MODEL + col] = acc[nt][r] + bias;
    }
  }
}

// ---------------------------------------------------------------------------
// Host launcher
// ---------------------------------------------------------------------------
extern "C" void kernel_launch(void* const* d_in, const int* in_sizes, int n_in,
                              void* d_out, int out_size, void* d_ws,
                              size_t ws_size, hipStream_t stream) {
  (void)in_sizes; (void)n_in; (void)out_size; (void)ws_size;
  const float* x   = (const float*)d_in[0];
  const float* W_Q = (const float*)d_in[1];
  const float* W_K = (const float*)d_in[2];
  const float* W_V = (const float*)d_in[3];
  const float* W_O = (const float*)d_in[4];
  const float* b_Q = (const float*)d_in[5];
  const float* b_K = (const float*)d_in[6];
  const float* b_V = (const float*)d_in[7];
  const float* b_O = (const float*)d_in[8];
  float* out = (float*)d_out;

  char* ws = (char*)d_ws;
  size_t off = 0;
  bf16_t* xb  = (bf16_t*)(ws + off); off += (size_t)NTOK * D_MODEL * 2;        // 8 MB
  bf16_t* wqt = (bf16_t*)(ws + off); off += (size_t)N_HEADS * D_HEAD * D_MODEL * 2;
  bf16_t* wkt = (bf16_t*)(ws + off); off += (size_t)N_HEADS * D_HEAD * D_MODEL * 2;
  bf16_t* wvt = (bf16_t*)(ws + off); off += (size_t)N_HEADS * D_HEAD * D_MODEL * 2;
  bf16_t* wot = (bf16_t*)(ws + off); off += (size_t)D_MODEL * D_MODEL * 2;     // 2 MB
  bf16_t* Qb  = (bf16_t*)(ws + off); off += (size_t)NTOK * D_HEAD * N_HEADS * 2;
  bf16_t* Kb  = (bf16_t*)(ws + off); off += (size_t)NTOK * D_HEAD * N_HEADS * 2;
  bf16_t* Vt  = (bf16_t*)(ws + off); off += (size_t)NTOK * D_HEAD * N_HEADS * 2;
  bf16_t* Zb  = (bf16_t*)(ws + off); off += (size_t)NTOK * D_MODEL * 2;        // 8 MB

  // 1) precision conversion + weight transposes (make all B-tile loads contiguous)
  {
    int nx = NTOK * D_MODEL;
    convert_f32_bf16<<<(nx + 255) / 256, 256, 0, stream>>>(x, xb, nx);
    int nw = N_HEADS * D_MODEL * D_HEAD;
    transpose_convert<<<(nw + 255) / 256, 256, 0, stream>>>(W_Q, wqt, N_HEADS, D_MODEL, D_HEAD);
    transpose_convert<<<(nw + 255) / 256, 256, 0, stream>>>(W_K, wkt, N_HEADS, D_MODEL, D_HEAD);
    transpose_convert<<<(nw + 255) / 256, 256, 0, stream>>>(W_V, wvt, N_HEADS, D_MODEL, D_HEAD);
    int no = D_MODEL * D_MODEL;  // W_O viewed as [f=h*64+e][d]
    transpose_convert<<<(no + 255) / 256, 256, 0, stream>>>(W_O, wot, 1, D_MODEL, D_MODEL);
  }

  // 2) QKV projection (WMMA bf16, LDS-staged weights)
  {
    dim3 grid(NTOK / 64, N_HEADS, 3);
    qkv_kernel<<<grid, 128, 0, stream>>>(xb, wqt, wkt, wvt, b_Q, b_K, b_V,
                                         Qb, Kb, Vt);
  }

  // 3) causal flash attention (WMMA bf16, online softmax, pipelined loads)
  {
    dim3 grid(SEQ / 64, N_HEADS, BATCH);
    attn_kernel<<<grid, 128, 0, stream>>>(Qb, Kb, Vt, Zb);
  }

  // 4) output projection (WMMA bf16, LDS-staged weights, fp32 out)
  {
    dim3 grid(NTOK / 128, D_MODEL / 64);
    oproj_kernel<<<grid, 256, 0, stream>>>(Zb, wot, b_O, out);
  }
}
